// S5TrajectoryDecoder_22170621182555
// MI455X (gfx1250) — matrix operand order repaired and checked by reading, not compiled, not run on hardware
//
#include <hip/hip_runtime.h>
#include <math.h>

// ---- problem constants (match reference) ----
#define LSEQ  32768
#define DM    256
#define PP    128
#define NL    4
#define ODIM  64
#define CHUNK 128
#define NCH   (LSEQ / CHUNK)

typedef __attribute__((ext_vector_type(16))) __bf16 v16bf;
typedef __attribute__((ext_vector_type(8)))  float  v8f;

struct __align__(16) F4 { float x, y, z, w; };
union Frag { F4 f4[2]; v16bf v; };   // 32 bytes = one WMMA 16-bit A/B fragment

__device__ __forceinline__ unsigned short f2bf(float f) {
  unsigned u = __float_as_uint(f);
  u += 0x7fffu + ((u >> 16) & 1u);          // round-to-nearest-even
  return (unsigned short)(u >> 16);
}
__device__ __forceinline__ float bf2f(unsigned short h) {
  return __uint_as_float(((unsigned)h) << 16);
}

// ---------------------------------------------------------------------------
// Weight pre-swizzle into a WMMA B (32x16, 16-bit) fragment order, contiguous k per lane half (A uses the same):
// dword index d -> v = d&7, lane = (d>>3)&31, tile = d>>8,
// tile = kt*(N/16)+nt ; lane: n = nt*16 + (lane&15), K pair = kt*32 + (lane>>4)*16 + 2v
// ---------------------------------------------------------------------------
__device__ __forceinline__ void pack_decode(int d, int Ntot, int& k0, int& n) {
  int v    = d & 7;
  int lane = (d >> 3) & 31;
  int tile = d >> 8;
  int nt16 = Ntot >> 4;
  int nt   = tile % nt16;
  int kt   = tile / nt16;
  n  = nt * 16 + (lane & 15);
  k0 = kt * 32 + (lane >> 4) * 16 + 2 * v;
}

__global__ void k_pack_w(const float* __restrict__ src, unsigned* __restrict__ dst, int Ntot) {
  int d = blockIdx.x * 256 + threadIdx.x;
  int k0, n; pack_decode(d, Ntot, k0, n);
  unsigned lo = f2bf(src[k0 * Ntot + n]);
  unsigned hi = f2bf(src[(k0 + 1) * Ntot + n]);
  dst[d] = lo | (hi << 16);
}

// B_bar[p][h] = ((Lam_bar-1)/Lam)_p * (Bre+ i Bim)[p][h]; packed N = [re(0..127) | im(128..255)]
__global__ void k_pack_bbar(const float* __restrict__ Bre, const float* __restrict__ Bim,
                            const float* __restrict__ Bfac, unsigned* __restrict__ dst) {
  int d = blockIdx.x * 256 + threadIdx.x;
  int k0, n; pack_decode(d, DM, k0, n);
  int p = n & (PP - 1);
  int isIm = n >> 7;
  float fr = Bfac[2 * p], fi = Bfac[2 * p + 1];
  unsigned w = 0;
  for (int e = 0; e < 2; e++) {
    int h = k0 + e;
    float br = Bre[p * DM + h], bi = Bim[p * DM + h];
    float val = isIm ? (fr * bi + fi * br) : (fr * br - fi * bi);
    w |= ((unsigned)f2bf(val)) << (16 * e);
  }
  dst[d] = w;
}

// y = xs_re @ (2*Cre^T) + xs_im @ (-2*Cim^T): rows 0..127 = 2*Cre[n][k], 128..255 = -2*Cim[n][k-128]
__global__ void k_pack_c(const float* __restrict__ Cre, const float* __restrict__ Cim,
                         unsigned* __restrict__ dst) {
  int d = blockIdx.x * 256 + threadIdx.x;
  int k0, n; pack_decode(d, DM, k0, n);
  unsigned w = 0;
  for (int e = 0; e < 2; e++) {
    int k = k0 + e;
    float val = (k < PP) ? 2.f * Cre[n * PP + k] : -2.f * Cim[n * PP + (k - PP)];
    w |= ((unsigned)f2bf(val)) << (16 * e);
  }
  dst[d] = w;
}

// ZOH discretization + chunk-jump operator Lam_bar^CHUNK (closed form via exp)
__global__ void k_discretize(const float* __restrict__ Lre, const float* __restrict__ Lim,
                             const float* __restrict__ lstep,
                             float* __restrict__ LamBar, float* __restrict__ aS,
                             float* __restrict__ Bfac) {
  int i = blockIdx.x * PP + threadIdx.x;   // layer * P + p
  float st = expf(lstep[i]);
  float ar = Lre[i], ai = Lim[i];
  float er  = expf(ar * st);
  float lbr = er * cosf(ai * st), lbi = er * sinf(ai * st);
  LamBar[2 * i] = lbr; LamBar[2 * i + 1] = lbi;
  float eS = expf(ar * st * (float)CHUNK);
  aS[2 * i]     = eS * cosf(ai * st * (float)CHUNK);
  aS[2 * i + 1] = eS * sinf(ai * st * (float)CHUNK);
  float den = ar * ar + ai * ai;
  float nr = lbr - 1.f, ni = lbi;
  Bfac[2 * i]     = (nr * ar + ni * ai) / den;   // (Lam_bar-1)/Lam
  Bfac[2 * i + 1] = (ni * ar - nr * ai) / den;
}

// latent expand + leaky_relu, broadcast to all L rows
__global__ __launch_bounds__(256) void k_expand(const float* __restrict__ lat,
                                                const float* __restrict__ W,
                                                const float* __restrict__ b,
                                                float* __restrict__ X) {
  __shared__ float s[DM];
  int t = threadIdx.x;
  float acc = b[t];
  for (int k = 0; k < DM; k++) acc += lat[k] * W[k * DM + t];
  acc = acc > 0.f ? acc : 0.01f * acc;
  s[t] = acc;
  __syncthreads();
  int rows = LSEQ / gridDim.x;
  int r0 = blockIdx.x * rows;
  for (int r = 0; r < rows; r++) X[(size_t)(r0 + r) * DM + t] = s[t];
}

// rowwise LayerNorm (optionally + tanh-GELU) -> bf16 activations (WMMA A operand)
template <bool GELU>
__global__ __launch_bounds__(256) void k_ln(const float* __restrict__ X,
                                            const float* __restrict__ scale,
                                            const float* __restrict__ bias,
                                            unsigned short* __restrict__ H) {
  int row  = blockIdx.x * 8 + (threadIdx.x >> 5);
  int lane = threadIdx.x & 31;
  const float* xp = X + (size_t)row * DM + lane * 8;
  F4 a = *(const F4*)xp, c = *(const F4*)(xp + 4);
  float v[8] = {a.x, a.y, a.z, a.w, c.x, c.y, c.z, c.w};
  float s = 0.f, q = 0.f;
  #pragma unroll
  for (int i = 0; i < 8; i++) { s += v[i]; q += v[i] * v[i]; }
  for (int m = 16; m > 0; m >>= 1) { s += __shfl_xor(s, m, 32); q += __shfl_xor(q, m, 32); }
  float mu  = s * (1.f / DM);
  float var = q * (1.f / DM) - mu * mu;
  float rs  = rsqrtf(var + 1e-6f);
  #pragma unroll
  for (int i = 0; i < 8; i++) {
    int col = lane * 8 + i;
    float y = (v[i] - mu) * rs * scale[col] + bias[col];
    if (GELU) {
      float t = tanhf(0.7978845608f * (y + 0.044715f * y * y * y));
      y = 0.5f * y * (1.f + t);
    }
    H[(size_t)row * DM + col] = f2bf(y);
  }
}

__global__ void k_cvt(const float* __restrict__ X, unsigned short* __restrict__ H) {
  int i = blockIdx.x * 256 + threadIdx.x;
  H[i] = f2bf(X[i]);
}

// ---------------------------------------------------------------------------
// Chunked diagonal complex scan (A constant since resets==0):
//   phase A: per-chunk carry; phase B: carry scan with Lam_bar^CHUNK;
//   phase C: replay with prefix, emit packed bf16 xs [re | im]
// ---------------------------------------------------------------------------
__global__ __launch_bounds__(128) void k_scan_local(const float* __restrict__ Bu,
                                                    const float* __restrict__ LamBar,
                                                    float* __restrict__ carry) {
  int ch = blockIdx.x, p = threadIdx.x;
  float ar = LamBar[2 * p], ai = LamBar[2 * p + 1];
  float sr = 0.f, si = 0.f;
  const float* bp = Bu + (size_t)ch * CHUNK * DM;
  for (int t = 0; t < CHUNK; t++) {
    float br = bp[t * DM + p], bi = bp[t * DM + PP + p];
    float nr = ar * sr - ai * si + br;
    float ni = ar * si + ai * sr + bi;
    sr = nr; si = ni;
  }
  carry[ch * 2 * PP + 2 * p] = sr;
  carry[ch * 2 * PP + 2 * p + 1] = si;
}

__global__ __launch_bounds__(128) void k_scan_carry(const float* __restrict__ aS,
                                                    const float* __restrict__ carry,
                                                    float* __restrict__ prefix) {
  int p = threadIdx.x;
  float ar = aS[2 * p], ai = aS[2 * p + 1];
  float sr = 0.f, si = 0.f;
  for (int ch = 0; ch < NCH; ch++) {
    prefix[ch * 2 * PP + 2 * p] = sr;
    prefix[ch * 2 * PP + 2 * p + 1] = si;
    float cr = carry[ch * 2 * PP + 2 * p], ci = carry[ch * 2 * PP + 2 * p + 1];
    float nr = ar * sr - ai * si + cr;
    float ni = ar * si + ai * sr + ci;
    sr = nr; si = ni;
  }
}

__global__ __launch_bounds__(128) void k_scan_final(const float* __restrict__ Bu,
                                                    const float* __restrict__ LamBar,
                                                    const float* __restrict__ prefix,
                                                    unsigned short* __restrict__ XS) {
  int ch = blockIdx.x, p = threadIdx.x;
  float ar = LamBar[2 * p], ai = LamBar[2 * p + 1];
  float sr = prefix[ch * 2 * PP + 2 * p], si = prefix[ch * 2 * PP + 2 * p + 1];
  const float* bp = Bu + (size_t)ch * CHUNK * DM;
  unsigned short* xp = XS + (size_t)ch * CHUNK * DM;
  for (int t = 0; t < CHUNK; t++) {
    float br = bp[t * DM + p], bi = bp[t * DM + PP + p];
    float nr = ar * sr - ai * si + br;
    float ni = ar * si + ai * sr + bi;
    sr = nr; si = ni;
    xp[t * DM + p]      = f2bf(sr);
    xp[t * DM + PP + p] = f2bf(si);
  }
}

// ---------------------------------------------------------------------------
// Fused WMMA GEMM: 8 waves/block; wave tile = 16(M) x 64(N); K=256 in 8 steps of 32.
// Per K-step: batch-load ALL B fragments (one s_clause of global_load_b128),
// then issue the 4 (or 2x4) WMMAs -> memory latency overlapped with WMMA issue.
// MODE 0: Out = A@B1                  (Bu projection, fp32 out)
// MODE 1: Out = A@B1 + h*D            (C projection + D-term, fp32 out)
// MODE 2: Out = Skip + (A@B1+b1)*sigmoid(A@B2+b2)   (full GLU + residual)
// MODE 3: Out = A@B1 + b1             (final head, N=64)
// ---------------------------------------------------------------------------
template <int MODE>
__global__ __launch_bounds__(256) void k_gemm(const unsigned short* __restrict__ A,
                                              const unsigned* __restrict__ B1,
                                              const unsigned* __restrict__ B2,
                                              const float* __restrict__ bias1,
                                              const float* __restrict__ bias2,
                                              const unsigned short* __restrict__ Haux,
                                              const float* __restrict__ Dv,
                                              const float* __restrict__ Skip,
                                              float* __restrict__ Out,
                                              int Ntot) {
  const int lane   = threadIdx.x & 31;
  const int wave   = threadIdx.x >> 5;
  const int wavesN = Ntot >> 6;                 // 4 (N=256) or 1 (N=64)
  const int wavesM = 8 / wavesN;
  const int wm = wave / wavesN, wn = wave % wavesN;
  const int m0 = blockIdx.x * (16 * wavesM) + wm * 16;
  const int n0 = wn * 64;
  const int half = lane >> 4;
  const int arow = m0 + (lane & 15);
  const int nt16 = Ntot >> 4;

  // hoisted bases: per K-step deltas become immediate offsets
  const unsigned short* aBase = A + (size_t)arow * DM + half * 16;     // +32 elems / K-step
  const size_t bStepDw = (size_t)nt16 << 8;                            // dwords per kt step
  const unsigned* b1Base = B1 + ((((size_t)(n0 >> 4)) << 5) + lane) * 8;
  const unsigned* b2Base = (MODE == 2) ? B2 + ((((size_t)(n0 >> 4)) << 5) + lane) * 8 : nullptr;

  v8f acc[4]  = {};
  v8f acc2[4] = {};

  #pragma unroll
  for (int kt = 0; kt < 8; kt++) {
    Frag af;
    const unsigned short* ap = aBase + kt * 32;
    af.f4[0] = *(const F4*)ap;
    af.f4[1] = *(const F4*)(ap + 8);

    Frag bf[4];
    #pragma unroll
    for (int nt = 0; nt < 4; nt++) {
      const unsigned* bp = b1Base + kt * bStepDw + ((size_t)nt << 8);
      bf[nt].f4[0] = *(const F4*)bp;
      bf[nt].f4[1] = *(const F4*)(bp + 4);
    }
    #pragma unroll
    for (int nt = 0; nt < 4; nt++)
      acc[nt] = __builtin_amdgcn_wmma_f32_16x16x32_bf16(false, af.v, false, bf[nt].v,
                                                        (short)0, acc[nt], false, false);
    if (MODE == 2) {
      Frag bg[4];
      #pragma unroll
      for (int nt = 0; nt < 4; nt++) {
        const unsigned* bq = b2Base + kt * bStepDw + ((size_t)nt << 8);
        bg[nt].f4[0] = *(const F4*)bq;
        bg[nt].f4[1] = *(const F4*)(bq + 4);
      }
      #pragma unroll
      for (int nt = 0; nt < 4; nt++)
        acc2[nt] = __builtin_amdgcn_wmma_f32_16x16x32_bf16(false, af.v, false, bg[nt].v,
                                                           (short)0, acc2[nt], false, false);
    }
  }

  #pragma unroll
  for (int nt = 0; nt < 4; nt++) {
    int col = n0 + nt * 16 + (lane & 15);
    #pragma unroll
    for (int r = 0; r < 8; r++) {
      int row = m0 + r + half * 8;
      float v = acc[nt][r];
      if (MODE == 0) {
        Out[(size_t)row * Ntot + col] = v;
      } else if (MODE == 1) {
        Out[(size_t)row * Ntot + col] = v + bf2f(Haux[(size_t)row * DM + col]) * Dv[col];
      } else if (MODE == 2) {
        float a1 = v + bias1[col];
        float a2 = acc2[nt][r] + bias2[col];
        Out[(size_t)row * Ntot + col] =
            Skip[(size_t)row * DM + col] + a1 * (1.f / (1.f + expf(-a2)));
      } else {
        Out[(size_t)row * Ntot + col] = v + bias1[col];
      }
    }
  }
}

// ---------------------------------------------------------------------------
extern "C" void kernel_launch(void* const* d_in, const int* in_sizes, int n_in,
                              void* d_out, int out_size, void* d_ws, size_t ws_size,
                              hipStream_t stream) {
  (void)in_sizes; (void)n_in; (void)out_size; (void)ws_size;
  const float* latent   = (const float*)d_in[0];
  const float* W_expand = (const float*)d_in[1];
  const float* b_expand = (const float*)d_in[2];
  const float* n_scale  = (const float*)d_in[3];
  const float* n_bias   = (const float*)d_in[4];
  const float* Lre      = (const float*)d_in[5];
  const float* Lim      = (const float*)d_in[6];
  const float* Bre      = (const float*)d_in[7];
  const float* Bim      = (const float*)d_in[8];
  const float* Cre      = (const float*)d_in[9];
  const float* Cim      = (const float*)d_in[10];
  const float* Dvec     = (const float*)d_in[11];
  const float* lstep    = (const float*)d_in[12];
  const float* W1       = (const float*)d_in[13];
  const float* b1       = (const float*)d_in[14];
  const float* W2       = (const float*)d_in[15];
  const float* b2       = (const float*)d_in[16];
  const float* Wout     = (const float*)d_in[17];
  const float* bout     = (const float*)d_in[18];

  char* ws = (char*)d_ws;
  float*          X    = (float*)ws;           ws += (size_t)LSEQ * DM * 4;  // residual stream
  float*          BU   = (float*)ws;           ws += (size_t)LSEQ * DM * 4;  // Bu, reused as Ypre
  unsigned short* H    = (unsigned short*)ws;  ws += (size_t)LSEQ * DM * 2;  // h (bf16), reused as g / xb
  unsigned short* XS   = (unsigned short*)ws;  ws += (size_t)LSEQ * DM * 2;  // packed scan states (bf16)
  float* carry   = (float*)ws;  ws += (size_t)NCH * 2 * PP * 4;
  float* prefix  = (float*)ws;  ws += (size_t)NCH * 2 * PP * 4;
  float* LamBar  = (float*)ws;  ws += (size_t)NL * 2 * PP * 4;
  float* aS      = (float*)ws;  ws += (size_t)NL * 2 * PP * 4;
  float* Bfac    = (float*)ws;  ws += (size_t)NL * 2 * PP * 4;
  unsigned* Bbp  = (unsigned*)ws; ws += (size_t)NL * (DM * DM / 2) * 4;
  unsigned* Cp   = (unsigned*)ws; ws += (size_t)NL * (DM * DM / 2) * 4;
  unsigned* W1p  = (unsigned*)ws; ws += (size_t)NL * (DM * DM / 2) * 4;
  unsigned* W2p  = (unsigned*)ws; ws += (size_t)NL * (DM * DM / 2) * 4;
  unsigned* Wop  = (unsigned*)ws; ws += (size_t)(DM * ODIM / 2) * 4;

  // stage 0: expand latent + broadcast, discretize, pack weights to WMMA-B layout
  k_expand<<<256, 256, 0, stream>>>(latent, W_expand, b_expand, X);
  k_discretize<<<NL, 128, 0, stream>>>(Lre, Lim, lstep, LamBar, aS, Bfac);
  for (int l = 0; l < NL; l++) {
    k_pack_bbar<<<128, 256, 0, stream>>>(Bre + (size_t)l * PP * DM, Bim + (size_t)l * PP * DM,
                                         Bfac + l * 2 * PP, Bbp + (size_t)l * (DM * DM / 2));
    k_pack_c<<<128, 256, 0, stream>>>(Cre + (size_t)l * DM * PP, Cim + (size_t)l * DM * PP,
                                      Cp + (size_t)l * (DM * DM / 2));
    k_pack_w<<<128, 256, 0, stream>>>(W1 + (size_t)l * DM * DM, W1p + (size_t)l * (DM * DM / 2), DM);
    k_pack_w<<<128, 256, 0, stream>>>(W2 + (size_t)l * DM * DM, W2p + (size_t)l * (DM * DM / 2), DM);
  }
  k_pack_w<<<(DM * ODIM / 2) / 256, 256, 0, stream>>>(Wout, Wop, ODIM);

  // stage 1: four S5 layers
  for (int l = 0; l < NL; l++) {
    const unsigned wOff = (unsigned)l * (DM * DM / 2);
    k_ln<false><<<LSEQ / 8, 256, 0, stream>>>(X, n_scale + l * DM, n_bias + l * DM, H);
    k_gemm<0><<<LSEQ / 32, 256, 0, stream>>>(H, Bbp + wOff, nullptr, nullptr, nullptr,
                                             nullptr, nullptr, nullptr, BU, DM);
    k_scan_local<<<NCH, 128, 0, stream>>>(BU, LamBar + l * 2 * PP, carry);
    k_scan_carry<<<1, 128, 0, stream>>>(aS + l * 2 * PP, carry, prefix);
    k_scan_final<<<NCH, 128, 0, stream>>>(BU, LamBar + l * 2 * PP, prefix, XS);
    k_gemm<1><<<LSEQ / 32, 256, 0, stream>>>(XS, Cp + wOff, nullptr, nullptr, nullptr,
                                             H, Dvec + l * DM, nullptr, BU /*Ypre*/, DM);
    k_ln<true><<<LSEQ / 8, 256, 0, stream>>>(BU, n_scale + l * DM, n_bias + l * DM, H /*g*/);
    k_gemm<2><<<LSEQ / 32, 256, 0, stream>>>(H, W1p + wOff, W2p + wOff,
                                             b1 + l * DM, b2 + l * DM,
                                             nullptr, nullptr, X /*skip*/, X /*out*/, DM);
  }

  // stage 2: output head
  k_cvt<<<(LSEQ * DM) / 256, 256, 0, stream>>>(X, H);
  k_gemm<3><<<LSEQ / 128, 256, 0, stream>>>(H, Wop, nullptr, bout, nullptr,
                                            nullptr, nullptr, nullptr, (float*)d_out, ODIM);
}
